// Attention_14740327760160
// MI455X (gfx1250) — matrix operand order repaired and checked by reading, not compiled, not run on hardware
//
#include <hip/hip_runtime.h>
#include <hip/hip_bf16.h>

typedef __bf16 bf16_t;
typedef __attribute__((ext_vector_type(16))) __bf16 v16bf;
typedef __attribute__((ext_vector_type(8)))  __bf16 v8bf;
typedef __attribute__((ext_vector_type(4)))  __bf16 v4bf;
typedef __attribute__((ext_vector_type(4)))  float  v4f;
typedef __attribute__((ext_vector_type(8)))  float  v8f;

#define WAVES_PER_BLOCK 4

// ---------- helpers ----------

__device__ inline v16bf load_frag(const bf16_t* p0, const bf16_t* p1) {
  union { v16bf v; v8bf h[2]; } u;
  u.h[0] = *(const v8bf*)p0;   // 16B load
  u.h[1] = *(const v8bf*)p1;   // 16B load
  return u.v;
}

__device__ inline v8f wmma_bf16(v16bf a, v16bf b, v8f c) {
  // D = A(16x32 bf16) * B(32x16 bf16) + C(16x16 f32)
  return __builtin_amdgcn_wmma_f32_16x16x32_bf16(
      /*neg_a=*/false, a, /*neg_b=*/false, b,
      /*c_mod=*/(short)0, c, /*reuse_a=*/false, /*reuse_b=*/false);
}

__device__ inline v8f vzero8() {
  v8f z = {0.f,0.f,0.f,0.f,0.f,0.f,0.f,0.f};
  return z;
}

// LDS byte offset of a generic pointer into __shared__ (addrspace(3) is 32-bit)
__device__ inline unsigned lds_off(const void* p) {
  return (unsigned)(unsigned long long)(__attribute__((address_space(3))) const void*)p;
}

// Async 16B copy global -> LDS (ASYNCcnt-tracked, per-lane addresses)
__device__ inline void async_ld16(unsigned lds_byte_off, const void* gptr) {
  unsigned long long ga = (unsigned long long)gptr;
  asm volatile("global_load_async_to_lds_b128 %0, %1, off"
               :: "v"(lds_byte_off), "v"(ga) : "memory");
}

// ---------- kernel 0: pack weights f32[1024][N] -> bf16 tiles [jb][kb][col16][kk32] ----------

__global__ void pack_w_kernel(const float* __restrict__ w, bf16_t* __restrict__ out,
                              int Ncols) {                 // K fixed at 1024 (KB = 32)
  int idx = blockIdx.x * blockDim.x + threadIdx.x;
  int total = 1024 * Ncols;
  if (idx >= total) return;
  int kk  = idx & 31;
  int col = (idx >> 5) & 15;
  int t   = idx >> 9;            // jb*32 + kb
  int kb  = t & 31;
  int jb  = t >> 5;
  int k = kb * 32 + kk;
  int j = jb * 16 + col;
  out[idx] = (bf16_t)w[(size_t)k * Ncols + j];
}

// ---------- kernel 1: cast x f32 -> bf16 (4 elements/thread) ----------

__global__ void cast_f32_bf16_kernel(const float* __restrict__ in, bf16_t* __restrict__ out, int n4) {
  int i = blockIdx.x * blockDim.x + threadIdx.x;
  if (i >= n4) return;
  v4f x = *(const v4f*)(in + (size_t)i * 4);
  v4bf y;
  y.x = (bf16_t)x.x; y.y = (bf16_t)x.y; y.z = (bf16_t)x.z; y.w = (bf16_t)x.w;
  *(v4bf*)(out + (size_t)i * 4) = y;
}

// ---------- kernel 2: qkv = x @ w_qkv ; 32x64 tile/wave; scatter q*scale, k, vT ----------

__global__ __launch_bounds__(128) void gemm_qkv_kernel(
    const bf16_t* __restrict__ xb,   // [8192][1024] bf16 row-major
    const bf16_t* __restrict__ wp,   // packed, Ncols=3072, K=1024
    bf16_t* __restrict__ qf,         // [64][2048][64]
    bf16_t* __restrict__ kf,         // [64][2048][64]
    bf16_t* __restrict__ vT)         // [64][64][2048]
{
  const int lane = threadIdx.x & 31;
  const int l15  = lane & 15;
  const int half = lane >> 4;
  const int wid  = blockIdx.x * WAVES_PER_BLOCK + (threadIdx.x >> 5);
  const int CG   = 48;               // 3072/64
  const int cg   = wid % CG;
  const int rt   = wid / CG;         // 0..255 (32 rows each)
  const int row0 = rt * 32;
  const int j0   = cg * 64;
  const int KB   = 32;               // 1024/32

  v8f acc[2][4];
  #pragma unroll
  for (int s = 0; s < 2; ++s)
    #pragma unroll
    for (int t = 0; t < 4; ++t) acc[s][t] = vzero8();

  const bf16_t* arow0 = xb + (size_t)(row0 + l15) * 1024 + half * 16;
  const bf16_t* arow1 = arow0 + (size_t)16 * 1024;

  // prologue: A fragments for kc=0
  v16bf a0 = load_frag(arow0, arow0 + 8);
  v16bf a1 = load_frag(arow1, arow1 + 8);

  for (int kc = 0; kc < 32; ++kc) {
    // batch all B-fragment loads for this K-step
    const bf16_t* bp0 = wp + ((size_t)(((j0 >> 4) + 0) * KB + kc) * 512) + l15 * 32 + half * 16;
    const bf16_t* bp1 = wp + ((size_t)(((j0 >> 4) + 1) * KB + kc) * 512) + l15 * 32 + half * 16;
    const bf16_t* bp2 = wp + ((size_t)(((j0 >> 4) + 2) * KB + kc) * 512) + l15 * 32 + half * 16;
    const bf16_t* bp3 = wp + ((size_t)(((j0 >> 4) + 3) * KB + kc) * 512) + l15 * 32 + half * 16;
    v16bf b0 = load_frag(bp0, bp0 + 8);
    v16bf b1 = load_frag(bp1, bp1 + 8);
    v16bf b2 = load_frag(bp2, bp2 + 8);
    v16bf b3 = load_frag(bp3, bp3 + 8);

    // software prefetch of next A fragments (overlaps with WMMAs below)
    int kn = (kc + 1 < 32) ? (kc + 1) : kc;
    v16bf a0n = load_frag(arow0 + kn * 32, arow0 + kn * 32 + 8);
    v16bf a1n = load_frag(arow1 + kn * 32, arow1 + kn * 32 + 8);

    acc[0][0] = wmma_bf16(a0, b0, acc[0][0]);
    acc[1][0] = wmma_bf16(a1, b0, acc[1][0]);
    acc[0][1] = wmma_bf16(a0, b1, acc[0][1]);
    acc[1][1] = wmma_bf16(a1, b1, acc[1][1]);
    acc[0][2] = wmma_bf16(a0, b2, acc[0][2]);
    acc[1][2] = wmma_bf16(a1, b2, acc[1][2]);
    acc[0][3] = wmma_bf16(a0, b3, acc[0][3]);
    acc[1][3] = wmma_bf16(a1, b3, acc[1][3]);

    a0 = a0n; a1 = a1n;
  }

  const int h     = (j0 & 1023) >> 6;  // uniform per wave (64-aligned col group)
  const int which = j0 >> 10;          // 0=q, 1=k, 2=v
  #pragma unroll
  for (int s = 0; s < 2; ++s) {
    #pragma unroll
    for (int t = 0; t < 4; ++t) {
      int d = (t * 16) + l15;          // d within head
      #pragma unroll
      for (int r = 0; r < 8; ++r) {
        int ng = row0 + s * 16 + r + 8 * half;   // flat row over B*N
        int b_ = ng >> 11, n = ng & 2047;
        float v = acc[s][t][r];
        if (which == 0) {
          qf[((size_t)(b_ * 16 + h) * 2048 + n) * 64 + d] = (bf16_t)(v * 0.125f); // fold D^-0.5
        } else if (which == 1) {
          kf[((size_t)(b_ * 16 + h) * 2048 + n) * 64 + d] = (bf16_t)v;
        } else {
          vT[((size_t)(b_ * 16 + h) * 64 + d) * 2048 + n] = (bf16_t)v;
        }
      }
    }
  }
}

// ---------- kernel 3: flash attention ----------
// Block = 4 waves, same (b,h), 4 consecutive 16-query tiles. 64-key K/V blocks
// staged into LDS with async global->LDS copies, double-buffered (ping-pong).

__global__ __launch_bounds__(128) void attn_kernel(
    const bf16_t* __restrict__ qf,   // [64][2048][64] (pre-scaled)
    const bf16_t* __restrict__ kf,   // [64][2048][64]
    const bf16_t* __restrict__ vT,   // [64][64][2048]
    bf16_t* __restrict__ ob)         // [8192][1024]  ([B,N,C] bf16)
{
  __shared__ bf16_t kbuf[2][64 * 64];               // 2 x 8KB: [key][d]
  __shared__ bf16_t vbuf[2][64 * 64];               // 2 x 8KB: [d][key]
  __shared__ bf16_t pbuf[WAVES_PER_BLOCK][16 * 64]; // per-wave P transpose bounce (2KB)

  const int lane  = threadIdx.x & 31;
  const int l15   = lane & 15;
  const int half  = lane >> 4;
  const int wslot = threadIdx.x >> 5;
  const int bh    = blockIdx.x >> 5;            // 0..63
  const int qt    = (blockIdx.x & 31) * 4 + wslot;

  const bf16_t* qp = qf + (size_t)bh * 2048 * 64;
  const bf16_t* kp = kf + (size_t)bh * 2048 * 64;
  const bf16_t* vp = vT + (size_t)bh * 64 * 2048;
  bf16_t* pl = pbuf[wslot];

  // Q A-fragments, d-chunks [0,32) and [32,64)
  const bf16_t* qrow = qp + (size_t)(qt * 16 + l15) * 64 + half * 16;
  v16bf aq0 = load_frag(qrow,      qrow + 8);
  v16bf aq1 = load_frag(qrow + 32, qrow + 40);

  v8f oacc[4];
  #pragma unroll
  for (int t = 0; t < 4; ++t) oacc[t] = vzero8();
  float m[8], l[8];
  #pragma unroll
  for (int r = 0; r < 8; ++r) { m[r] = -1e30f; l[r] = 0.f; }

  const float L2E = 1.44269504f;
  const int tid = threadIdx.x;

  // ---- stage 64-key block 0 into buffer 0 (8 async instructions per wave) ----
  {
    const char* kg = (const char*)kp;                    // key0 = 0: contiguous 8KB
    #pragma unroll
    for (int c = 0; c < 4; ++c) {
      int chunk = tid + c * 128;                         // 512 x 16B
      async_ld16(lds_off(&kbuf[0][0]) + chunk * 16, kg + chunk * 16);
    }
    #pragma unroll
    for (int c = 0; c < 4; ++c) {
      int chunk = tid + c * 128;                         // d = chunk>>3, seg = chunk&7
      int d = chunk >> 3, seg = chunk & 7;
      const char* g = (const char*)vp + ((size_t)d * 2048) * 2 + seg * 16;
      async_ld16(lds_off(&vbuf[0][0]) + chunk * 16, g);
    }
  }

  for (int kc = 0; kc < 32; ++kc) {
    const int cur = kc & 1;
    const int nxt = cur ^ 1;

    if (kc + 1 < 32) {
      __syncthreads();   // everyone done reading buf[nxt] (from iteration kc-1)
      const int key1 = (kc + 1) * 64;
      const char* kg = (const char*)(kp + (size_t)key1 * 64);
      #pragma unroll
      for (int c = 0; c < 4; ++c) {
        int chunk = tid + c * 128;
        async_ld16(lds_off(&kbuf[nxt][0]) + chunk * 16, kg + chunk * 16);
      }
      #pragma unroll
      for (int c = 0; c < 4; ++c) {
        int chunk = tid + c * 128;
        int d = chunk >> 3, seg = chunk & 7;
        const char* g = (const char*)vp + ((size_t)d * 2048 + key1) * 2 + seg * 16;
        async_ld16(lds_off(&vbuf[nxt][0]) + chunk * 16, g);
      }
      // 8 newly issued ops outstanding; wait for the previous block's 8 to land
      asm volatile("s_wait_asynccnt 0x8" ::: "memory");
    } else {
      asm volatile("s_wait_asynccnt 0x0" ::: "memory");
    }
    __syncthreads();     // buf[cur] fully staged by all waves

    const bf16_t* kb = kbuf[cur];
    const bf16_t* vb = vbuf[cur];

    // ---- S = Q K^T for 64 keys (four 16x16 tiles), K fragments from LDS ----
    v8f s[4];
    #pragma unroll
    for (int j = 0; j < 4; ++j) {
      const bf16_t* kj = kb + (size_t)(j * 16 + l15) * 64 + half * 16;
      v8f sj = vzero8();
      sj = wmma_bf16(aq0, load_frag(kj,      kj + 8),  sj);   // d chunk 0
      sj = wmma_bf16(aq1, load_frag(kj + 32, kj + 40), sj);   // d chunk 1
      s[j] = sj;
    }

    // ---- online softmax over the 64 new keys (rows in half-wave layout) ----
    #pragma unroll
    for (int r = 0; r < 8; ++r) {
      float x0 = s[0][r], x1 = s[1][r], x2 = s[2][r], x3 = s[3][r];
      float mx = fmaxf(fmaxf(x0, x1), fmaxf(x2, x3));
      mx = fmaxf(mx, __shfl_xor(mx, 1));
      mx = fmaxf(mx, __shfl_xor(mx, 2));
      mx = fmaxf(mx, __shfl_xor(mx, 4));
      mx = fmaxf(mx, __shfl_xor(mx, 8));
      float mn = fmaxf(m[r], mx);
      float a_ = exp2f((m[r] - mn) * L2E);
      float p0 = exp2f((x0 - mn) * L2E);
      float p1 = exp2f((x1 - mn) * L2E);
      float p2 = exp2f((x2 - mn) * L2E);
      float p3 = exp2f((x3 - mn) * L2E);
      float rs = (p0 + p1) + (p2 + p3);
      rs += __shfl_xor(rs, 1);
      rs += __shfl_xor(rs, 2);
      rs += __shfl_xor(rs, 4);
      rs += __shfl_xor(rs, 8);
      l[r] = l[r] * a_ + rs;
      m[r] = mn;
      oacc[0][r] *= a_; oacc[1][r] *= a_; oacc[2][r] *= a_; oacc[3][r] *= a_;
      // P (D-layout) -> LDS row-major [16 q][64 k] for A-fragment reload
      int row = r + 8 * half;
      pl[row * 64 + l15]      = (bf16_t)p0;
      pl[row * 64 + 16 + l15] = (bf16_t)p1;
      pl[row * 64 + 32 + l15] = (bf16_t)p2;
      pl[row * 64 + 48 + l15] = (bf16_t)p3;
    }
    asm volatile("s_wait_dscnt 0x0" ::: "memory");

    // ---- O += P V  (two 16x32 P A-fragments over the key halves) ----
    const bf16_t* pr = pl + l15 * 64 + half * 16;
    v16bf pa0 = load_frag(pr,      pr + 8);      // keys [0,32)
    v16bf pa1 = load_frag(pr + 32, pr + 40);     // keys [32,64)
    #pragma unroll
    for (int td = 0; td < 4; ++td) {
      const bf16_t* vv = vb + (size_t)(td * 16 + l15) * 64 + half * 16;
      oacc[td] = wmma_bf16(pa0, load_frag(vv, vv + 8), oacc[td]);
    }
    #pragma unroll
    for (int td = 0; td < 4; ++td) {
      const bf16_t* vv = vb + (size_t)(td * 16 + l15) * 64 + 32 + half * 16;
      oacc[td] = wmma_bf16(pa1, load_frag(vv, vv + 8), oacc[td]);
    }
  }

  // ---- normalize and store O as bf16 [B,N,C] ----
  const int b_ = bh >> 4, h = bh & 15;
  #pragma unroll
  for (int r = 0; r < 8; ++r) {
    float invl = 1.0f / l[r];
    int n = qt * 16 + r + 8 * half;
    #pragma unroll
    for (int td = 0; td < 4; ++td) {
      int c = h * 64 + td * 16 + l15;
      ob[((size_t)(b_ * 2048 + n)) * 1024 + c] = (bf16_t)(oacc[td][r] * invl);
    }
  }
}

// ---------- kernel 4: out = o @ w_proj + b_proj (f32 out), 32x64 tile/wave ----------

__global__ __launch_bounds__(128) void gemm_proj_kernel(
    const bf16_t* __restrict__ ab,   // [8192][1024] bf16
    const bf16_t* __restrict__ wp,   // packed, Ncols=1024, K=1024
    const float* __restrict__ bias,  // [1024]
    float* __restrict__ out)         // [8192][1024] f32
{
  const int lane = threadIdx.x & 31;
  const int l15  = lane & 15;
  const int half = lane >> 4;
  const int wid  = blockIdx.x * WAVES_PER_BLOCK + (threadIdx.x >> 5);
  const int CG   = 16;               // 1024/64
  const int cg   = wid % CG;
  const int rt   = wid / CG;         // 0..255
  const int row0 = rt * 32;
  const int j0   = cg * 64;
  const int KB   = 32;

  v8f acc[2][4];
  #pragma unroll
  for (int s = 0; s < 2; ++s)
    #pragma unroll
    for (int t = 0; t < 4; ++t) acc[s][t] = vzero8();

  const bf16_t* arow0 = ab + (size_t)(row0 + l15) * 1024 + half * 16;
  const bf16_t* arow1 = arow0 + (size_t)16 * 1024;

  v16bf a0 = load_frag(arow0, arow0 + 8);
  v16bf a1 = load_frag(arow1, arow1 + 8);

  for (int kc = 0; kc < 32; ++kc) {
    const bf16_t* bp0 = wp + ((size_t)(((j0 >> 4) + 0) * KB + kc) * 512) + l15 * 32 + half * 16;
    const bf16_t* bp1 = wp + ((size_t)(((j0 >> 4) + 1) * KB + kc) * 512) + l15 * 32 + half * 16;
    const bf16_t* bp2 = wp + ((size_t)(((j0 >> 4) + 2) * KB + kc) * 512) + l15 * 32 + half * 16;
    const bf16_t* bp3 = wp + ((size_t)(((j0 >> 4) + 3) * KB + kc) * 512) + l15 * 32 + half * 16;
    v16bf b0 = load_frag(bp0, bp0 + 8);
    v16bf b1 = load_frag(bp1, bp1 + 8);
    v16bf b2 = load_frag(bp2, bp2 + 8);
    v16bf b3 = load_frag(bp3, bp3 + 8);

    int kn = (kc + 1 < 32) ? (kc + 1) : kc;
    v16bf a0n = load_frag(arow0 + kn * 32, arow0 + kn * 32 + 8);
    v16bf a1n = load_frag(arow1 + kn * 32, arow1 + kn * 32 + 8);

    acc[0][0] = wmma_bf16(a0, b0, acc[0][0]);
    acc[1][0] = wmma_bf16(a1, b0, acc[1][0]);
    acc[0][1] = wmma_bf16(a0, b1, acc[0][1]);
    acc[1][1] = wmma_bf16(a1, b1, acc[1][1]);
    acc[0][2] = wmma_bf16(a0, b2, acc[0][2]);
    acc[1][2] = wmma_bf16(a1, b2, acc[1][2]);
    acc[0][3] = wmma_bf16(a0, b3, acc[0][3]);
    acc[1][3] = wmma_bf16(a1, b3, acc[1][3]);

    a0 = a0n; a1 = a1n;
  }

  #pragma unroll
  for (int s = 0; s < 2; ++s) {
    #pragma unroll
    for (int t = 0; t < 4; ++t) {
      int j = j0 + t * 16 + l15;
      float bj = bias[j];
      #pragma unroll
      for (int r = 0; r < 8; ++r) {
        int ng = row0 + s * 16 + r + 8 * half;
        out[(size_t)ng * 1024 + j] = acc[s][t][r] + bj;
      }
    }
  }
}

// ---------- launcher ----------

extern "C" void kernel_launch(void* const* d_in, const int* in_sizes, int n_in,
                              void* d_out, int out_size, void* d_ws, size_t ws_size,
                              hipStream_t stream) {
  const float* x      = (const float*)d_in[0];   // [4,2048,1024]
  const float* w_qkv  = (const float*)d_in[1];   // [1024,3072]
  const float* w_proj = (const float*)d_in[2];   // [1024,1024]
  const float* b_proj = (const float*)d_in[3];   // [1024]
  float* out = (float*)d_out;                    // [4,2048,1024]

  char* ws = (char*)d_ws;
  size_t off = 0;
  auto alloc = [&](size_t bytes) -> void* {
    void* p = ws + off;
    off = (off + bytes + 255) & ~(size_t)255;
    return p;
  };

  bf16_t* xb  = (bf16_t*)alloc((size_t)8192 * 1024 * 2);   // 16 MB
  bf16_t* wqp = (bf16_t*)alloc((size_t)1024 * 3072 * 2);   // 6 MB
  bf16_t* wpp = (bf16_t*)alloc((size_t)1024 * 1024 * 2);   // 2 MB
  bf16_t* qf  = (bf16_t*)alloc((size_t)64 * 2048 * 64 * 2);// 16 MB
  bf16_t* kf  = (bf16_t*)alloc((size_t)64 * 2048 * 64 * 2);// 16 MB
  bf16_t* vT  = (bf16_t*)alloc((size_t)64 * 64 * 2048 * 2);// 16 MB
  bf16_t* ob  = (bf16_t*)alloc((size_t)8192 * 1024 * 2);   // 16 MB
  (void)ws_size;

  pack_w_kernel<<<(1024 * 3072 + 255) / 256, 256, 0, stream>>>(w_qkv, wqp, 3072);
  pack_w_kernel<<<(1024 * 1024 + 255) / 256, 256, 0, stream>>>(w_proj, wpp, 1024);
  cast_f32_bf16_kernel<<<(8192 * 1024 / 4 + 255) / 256, 256, 0, stream>>>(x, xb, 8192 * 1024 / 4);

  gemm_qkv_kernel<<<(256 * 48) / WAVES_PER_BLOCK, 32 * WAVES_PER_BLOCK, 0, stream>>>(xb, wqp, qf, kf, vT);
  attn_kernel<<<64 * 32, 32 * WAVES_PER_BLOCK, 0, stream>>>(qf, kf, vT, ob);
  gemm_proj_kernel<<<(256 * 16) / WAVES_PER_BLOCK, 32 * WAVES_PER_BLOCK, 0, stream>>>(ob, wpp, b_proj, out);
}
